// CrossAttnBlock_24670292148862
// MI455X (gfx1250) — matrix operand order repaired and checked by reading, not compiled, not run on hardware
//
#include <hip/hip_runtime.h>
#include <hip/hip_bf16.h>
#include <stdint.h>

// ---------------------------------------------------------------------------
// Problem constants (from the reference)
// ---------------------------------------------------------------------------
#define BT    192
#define NQ    64
#define NCTX  1024
#define CDIM  384
#define NH    8
#define DH    48          // CDIM / NH
#define HID   1536
#define NROWS_X  (BT * NQ)      // 12288
#define NROWS_C  (BT * NCTX)    // 196608
#define ATTN_SCALE 0.14433756729740643f   // DH^-0.5

typedef unsigned short u16t;
typedef __attribute__((ext_vector_type(16))) __bf16 v16bf;
typedef __attribute__((ext_vector_type(8)))  float  v8f;

union Frag16 {
  v16bf v;
  u16t h[16];
  unsigned int w[8];
  uint4 q[2];
};
union AccF {
  v8f v;
  float f[8];
};

__device__ __forceinline__ u16t f32_to_bf16(float x) {
  unsigned int u = __float_as_uint(x);
  u += 0x7FFFu + ((u >> 16) & 1u);     // round-to-nearest-even
  return (u16t)(u >> 16);
}

// A-fragment (16x32 bf16) K index for 32-bit word v (0..7), lane half-group.
// ISA 7.12.2: lanes 0-15: V0..V3 -> K=0..7, V4..V7 -> K=16..23; lanes 16-31: +8
__device__ __forceinline__ int a_frag_k(int v, int hgrp) {
  return 2 * v + ((v >= 4) ? 8 : 0) + 8 * hgrp;
}

__device__ __forceinline__ v8f wmma_bf16(v16bf a, v16bf b, v8f c) {
  return __builtin_amdgcn_wmma_f32_16x16x32_bf16(
      /*neg_a=*/false, a, /*neg_b=*/false, b,
      /*c_mod=*/(short)0, c, /*reuse_a=*/false, /*reuse_b=*/false);
}

__device__ __forceinline__ float gelu_tanh(float x) {
  float x3 = x * x * x;
  float t = tanhf(0.7978845608028654f * (x + 0.044715f * x3));
  return 0.5f * x * (1.0f + t);
}

// ---------------------------------------------------------------------------
// CDNA5 async global->LDS copy (ASYNCcnt-tracked, no VGPR staging).
// GVS mode: mem = SADDR(64) + VADDR(32); LDS dest = VDST vgpr byte offset.
// LDS offset = low 32 bits of the flat address (LDS aperture: addr[31:0]).
// ---------------------------------------------------------------------------
__device__ __forceinline__ void async_load_b128(unsigned lds_off,
                                                unsigned glb_off,
                                                unsigned long long base) {
  asm volatile("global_load_async_to_lds_b128 %0, %1, %2"
               :: "v"(lds_off), "v"(glb_off), "s"(base)
               : "memory");
}
__device__ __forceinline__ void wait_async0() {
  asm volatile("s_wait_asynccnt 0x0" ::: "memory");
}

// ---------------------------------------------------------------------------
// f32 -> bf16 conversion (weights)
// ---------------------------------------------------------------------------
__global__ __launch_bounds__(256) void cvt_bf16_kernel(
    const float* __restrict__ in, u16t* __restrict__ out, int n) {
  int i = blockIdx.x * 256 + threadIdx.x;
  if (i < n) out[i] = f32_to_bf16(in[i]);
}

// ---------------------------------------------------------------------------
// Row LayerNorm over CDIM=384, optional affine, bf16 output. 1 wave32 per row.
// ---------------------------------------------------------------------------
__global__ __launch_bounds__(256) void ln_bf16_kernel(
    const float* __restrict__ in, const float* __restrict__ g,
    const float* __restrict__ bta, u16t* __restrict__ out,
    float eps, int rows) {
  const int row  = blockIdx.x * 8 + (threadIdx.x >> 5);
  const int lane = threadIdx.x & 31;
  if (row >= rows) return;
  const float* p = in + (size_t)row * CDIM;
  float x[12];
  float s = 0.f;
#pragma unroll
  for (int i = 0; i < 12; ++i) { x[i] = p[lane + 32 * i]; s += x[i]; }
#pragma unroll
  for (int off = 1; off < 32; off <<= 1) s += __shfl_xor(s, off, 32);
  const float mean = s * (1.0f / CDIM);
  float vs = 0.f;
#pragma unroll
  for (int i = 0; i < 12; ++i) { float d = x[i] - mean; vs += d * d; }
#pragma unroll
  for (int off = 1; off < 32; off <<= 1) vs += __shfl_xor(vs, off, 32);
  const float rstd = rsqrtf(vs * (1.0f / CDIM) + eps);
  u16t* o = out + (size_t)row * CDIM;
#pragma unroll
  for (int i = 0; i < 12; ++i) {
    int c = lane + 32 * i;
    float y = (x[i] - mean) * rstd;
    if (g != nullptr) y = y * g[c] + bta[c];
    o[c] = f32_to_bf16(y);
  }
}

// ---------------------------------------------------------------------------
// Tiled bf16 GEMM: C(MxN) = A(MxK) @ B(KxN), fp32 accumulate via WMMA.
// 256 threads = 8 waves; 128x128 block tile, BK=32, double-buffered LDS fed
// by async global->LDS DMA. Each wave computes a 32x64 register tile
// (2 A-frags x 4 B-frags = 8 WMMA per k-step).
// ---------------------------------------------------------------------------
enum { EPI_BF16 = 0, EPI_GELU_BF16 = 1, EPI_RESID_F32 = 2 };

template <int EPI>
__global__ __launch_bounds__(256) void gemm_bf16_kernel(
    const u16t* __restrict__ A, const u16t* __restrict__ B,
    const float* __restrict__ bias, const float* __restrict__ resid,
    u16t* __restrict__ out_bf, float* __restrict__ out_f,
    int M, int N, int K) {
  __shared__ u16t a_lds[2][128 * 32];   // 2 x 8 KB
  __shared__ u16t b_lds[2][32 * 128];   // 2 x 8 KB
  const int tid = threadIdx.x;
  const int w = tid >> 5;
  const int lane = tid & 31;
  const int lane16 = lane & 15;
  const int hgrp = lane >> 4;
  const int bm = blockIdx.y * 128;
  const int bn = blockIdx.x * 128;
  const int mi = (w & 3) * 32;          // wave m-offset within tile
  const int nj = (w >> 2) * 64;         // wave n-offset within tile

  // Cooperative copy mapping: 2 x b128 per thread per tile (4096 bf16 each).
  const int ar0 = tid >> 2,          ac0 = (tid & 3) * 8;
  const int ar1 = (tid + 256) >> 2,  ac1 = ((tid + 256) & 3) * 8;
  const int br0 = tid >> 4,          bc0 = (tid & 15) * 8;
  const int br1 = (tid + 256) >> 4,  bc1 = ((tid + 256) & 15) * 8;

  const unsigned ldsA0 = (unsigned)(size_t)&a_lds[0][ar0 * 32 + ac0];
  const unsigned ldsA1 = (unsigned)(size_t)&a_lds[0][ar1 * 32 + ac1];
  const unsigned ldsB0 = (unsigned)(size_t)&b_lds[0][br0 * 128 + bc0];
  const unsigned ldsB1 = (unsigned)(size_t)&b_lds[0][br1 * 128 + bc1];
  const unsigned long long aP = (unsigned long long)(size_t)A;
  const unsigned long long bP = (unsigned long long)(size_t)B;

  auto issue = [&](int buf, int k0) {
    const unsigned bofs = (unsigned)buf * (128u * 32u * 2u);  // 8 KB
    async_load_b128(ldsA0 + bofs, (unsigned)(((bm + ar0) * K + k0 + ac0) * 2), aP);
    async_load_b128(ldsA1 + bofs, (unsigned)(((bm + ar1) * K + k0 + ac1) * 2), aP);
    async_load_b128(ldsB0 + bofs, (unsigned)(((k0 + br0) * N + bn + bc0) * 2), bP);
    async_load_b128(ldsB1 + bofs, (unsigned)(((k0 + br1) * N + bn + bc1) * 2), bP);
  };

  AccF acc[2][4];
#pragma unroll
  for (int mf = 0; mf < 2; ++mf)
#pragma unroll
    for (int nf = 0; nf < 4; ++nf)
#pragma unroll
      for (int v = 0; v < 8; ++v) acc[mf][nf].f[v] = 0.f;

  const int ksteps = K >> 5;
  issue(0, 0);

  for (int s = 0; s < ksteps; ++s) {
    const int buf = s & 1;
    wait_async0();        // this wave's DMA into buf has landed in LDS
    __syncthreads();      // every wave's DMA has landed; buf^1 reads done
    if (s + 1 < ksteps) issue(buf ^ 1, (s + 1) * 32);

    Frag16 af[2];
#pragma unroll
    for (int mf = 0; mf < 2; ++mf)
#pragma unroll
      for (int v = 0; v < 8; ++v)
        af[mf].w[v] = *(const unsigned int*)&a_lds[buf][(mi + 16 * mf + lane16) * 32 +
                                                        a_frag_k(v, hgrp)];
    Frag16 bf4[4];
#pragma unroll
    for (int nf = 0; nf < 4; ++nf) {
#pragma unroll
      for (int e = 0; e < 16; ++e)
        bf4[nf].h[e] = b_lds[buf][(a_frag_k(e >> 1, hgrp) + (e & 1)) * 128 + nj + 16 * nf + lane16];
    }
#pragma unroll
    for (int mf = 0; mf < 2; ++mf)
#pragma unroll
      for (int nf = 0; nf < 4; ++nf)
        acc[mf][nf].v = wmma_bf16(af[mf].v, bf4[nf].v, acc[mf][nf].v);
  }

#pragma unroll
  for (int mf = 0; mf < 2; ++mf)
#pragma unroll
    for (int nf = 0; nf < 4; ++nf)
#pragma unroll
      for (int v = 0; v < 8; ++v) {
        const int row = bm + mi + 16 * mf + v + 8 * hgrp;
        const int col = bn + nj + 16 * nf + lane16;
        float x0 = acc[mf][nf].f[v] + bias[col];
        if (EPI == EPI_GELU_BF16) x0 = gelu_tanh(x0);
        if (EPI == EPI_RESID_F32) {
          size_t i0 = (size_t)row * N + col;
          out_f[i0] = x0 + resid[i0];
        } else {
          out_bf[(size_t)row * N + col] = f32_to_bf16(x0);
        }
      }
}

// ---------------------------------------------------------------------------
// Flash-style cross attention: one block (4 waves) per (batch, head).
// Q: 64x48 (padded K->64); stream 64-key K/V tiles; online softmax.
// Mask is a no-op (bias constant along softmax axis), so it is skipped.
// ---------------------------------------------------------------------------
__global__ __launch_bounds__(128) void attn_kernel(
    const u16t* __restrict__ q_bf, const u16t* __restrict__ kv_bf,
    u16t* __restrict__ o_bf) {
  __shared__ u16t q_lds[64 * 64];          // q rows x padded d
  __shared__ u16t kT[64 * 64];             // transposed K tile: [d_pad][key]
  __shared__ u16t v_lds[64 * 64];          // V tile: [key][d_pad]
  __shared__ u16t p_lds[4 * 16 * 64];      // per-wave P transpose buffer

  const int bh = blockIdx.x;
  const int b = bh >> 3;
  const int h = bh & 7;
  const int tid = threadIdx.x;
  const int w = tid >> 5;
  const int lane = tid & 31;
  const int lane16 = lane & 15;
  const int hgrp = lane >> 4;

  for (int i = tid; i < 64 * 64; i += 128) q_lds[i] = 0;
  for (int i = tid; i < 16 * 64; i += 128)
    kT[(48 + (i >> 6)) * 64 + (i & 63)] = 0;      // zero d-padding rows once
  __syncthreads();
  for (int i = tid; i < 64 * 48; i += 128) {
    int qq = i / 48, d = i % 48;
    q_lds[qq * 64 + d] = q_bf[(size_t)(b * NQ + qq) * CDIM + h * DH + d];
  }
  __syncthreads();

  // Per-wave Q A-fragments (rows 16w..16w+15, K padded to 64).
  Frag16 qa[2];
#pragma unroll
  for (int kc = 0; kc < 2; ++kc)
#pragma unroll
    for (int v = 0; v < 8; ++v)
      qa[kc].w[v] = *(const unsigned int*)&q_lds[(16 * w + lane16) * 64 +
                                                 32 * kc + a_frag_k(v, hgrp)];

  float m_run[8], l_run[8];
  AccF oacc[3];
#pragma unroll
  for (int v = 0; v < 8; ++v) { m_run[v] = -3.3e38f; l_run[v] = 0.f; }
#pragma unroll
  for (int c = 0; c < 3; ++c)
#pragma unroll
    for (int v = 0; v < 8; ++v) oacc[c].f[v] = 0.f;

  const size_t kvbase = (size_t)b * NCTX * (2 * CDIM);
  u16t* pw = &p_lds[w * 16 * 64];

  for (int kt = 0; kt < NCTX; kt += 64) {
    __syncthreads();
    for (int i = tid; i < 64 * 48; i += 128) {
      int key = i / 48, d = i % 48;
      size_t rb = kvbase + (size_t)(kt + key) * (2 * CDIM) + h * DH + d;
      kT[d * 64 + key] = kv_bf[rb];            // transpose K during load
      v_lds[key * 64 + d] = kv_bf[rb + CDIM];  // V row-major
    }
    __syncthreads();

    // sim tile: S (16 q-rows x 64 keys) per wave, fp32
    AccF s[4];
#pragma unroll
    for (int nc = 0; nc < 4; ++nc) {
#pragma unroll
      for (int v = 0; v < 8; ++v) s[nc].f[v] = 0.f;
#pragma unroll
      for (int kc = 0; kc < 2; ++kc) {
        Frag16 kb;   // B-frag: lane16 = N (key), element = K row (d index)
#pragma unroll
        for (int e = 0; e < 16; ++e)
          kb.h[e] = kT[(32 * kc + a_frag_k(e >> 1, hgrp) + (e & 1)) * 64 + 16 * nc + lane16];
        s[nc].v = wmma_bf16(qa[kc].v, kb.v, s[nc].v);
      }
    }

    // online softmax update (row index = v + 8*hgrp matches O accumulator)
#pragma unroll
    for (int v = 0; v < 8; ++v) {
      float mx = -3.3e38f;
#pragma unroll
      for (int nc = 0; nc < 4; ++nc) {
        s[nc].f[v] *= ATTN_SCALE;
        mx = fmaxf(mx, s[nc].f[v]);
      }
#pragma unroll
      for (int off = 1; off < 16; off <<= 1)
        mx = fmaxf(mx, __shfl_xor(mx, off, 32));
      const float mnew = fmaxf(m_run[v], mx);
      const float alpha = __expf(m_run[v] - mnew);
      float ls = 0.f;
#pragma unroll
      for (int nc = 0; nc < 4; ++nc) {
        float p = __expf(s[nc].f[v] - mnew);
        s[nc].f[v] = p;
        ls += p;
      }
#pragma unroll
      for (int off = 1; off < 16; off <<= 1)
        ls += __shfl_xor(ls, off, 32);
      l_run[v] = l_run[v] * alpha + ls;
      m_run[v] = mnew;
#pragma unroll
      for (int c = 0; c < 3; ++c) oacc[c].f[v] *= alpha;
      // C-layout -> LDS (row-major), re-read in A-layout (same-wave DS ops
      // are kept in order by hardware)
      const int prow = v + 8 * hgrp;
#pragma unroll
      for (int nc = 0; nc < 4; ++nc)
        pw[prow * 64 + nc * 16 + lane16] = f32_to_bf16(s[nc].f[v]);
    }

    // O += P @ V
#pragma unroll
    for (int kc = 0; kc < 2; ++kc) {
      Frag16 pa;
#pragma unroll
      for (int v = 0; v < 8; ++v)
        pa.w[v] = *(const unsigned int*)&pw[lane16 * 64 + 32 * kc +
                                           a_frag_k(v, hgrp)];
#pragma unroll
      for (int dc = 0; dc < 3; ++dc) {
        Frag16 vb;   // B-frag: lane16 = d, element = key row
#pragma unroll
        for (int e = 0; e < 16; ++e)
          vb.h[e] = v_lds[(32 * kc + a_frag_k(e >> 1, hgrp) + (e & 1)) * 64 + dc * 16 + lane16];
        oacc[dc].v = wmma_bf16(pa.v, vb.v, oacc[dc].v);
      }
    }
  }

#pragma unroll
  for (int v = 0; v < 8; ++v) {
    const float inv = 1.0f / l_run[v];
    const int qq = 16 * w + v + 8 * hgrp;
#pragma unroll
    for (int dc = 0; dc < 3; ++dc)
      o_bf[(size_t)(b * NQ + qq) * CDIM + h * DH + dc * 16 + lane16] =
          f32_to_bf16(oacc[dc].f[v] * inv);
  }
}

// ---------------------------------------------------------------------------
// Host-side orchestration
// ---------------------------------------------------------------------------
extern "C" void kernel_launch(void* const* d_in, const int* in_sizes, int n_in,
                              void* d_out, int out_size, void* d_ws,
                              size_t ws_size, hipStream_t stream) {
  (void)in_sizes; (void)n_in; (void)out_size; (void)ws_size;
  const float* x     = (const float*)d_in[0];
  const float* ctx   = (const float*)d_in[1];
  // d_in[2] = mask: bias is constant along the softmax axis -> no-op, skipped.
  const float* Wq    = (const float*)d_in[3];
  const float* bq    = (const float*)d_in[4];
  const float* Wkv   = (const float*)d_in[5];
  const float* bkv   = (const float*)d_in[6];
  const float* Wo    = (const float*)d_in[7];
  const float* bo    = (const float*)d_in[8];
  const float* g_ctx = (const float*)d_in[9];
  const float* b_ctx = (const float*)d_in[10];
  const float* W1    = (const float*)d_in[11];
  const float* b1p   = (const float*)d_in[12];
  const float* W2    = (const float*)d_in[13];
  const float* b2p   = (const float*)d_in[14];
  float* out = (float*)d_out;

  char* ws = (char*)d_ws;
  size_t off = 0;
  auto alloc = [&](size_t bytes) -> char* {
    off = (off + 255) & ~(size_t)255;
    char* p = ws + off;
    off += bytes;
    return p;
  };

  u16t* xn_bf  = (u16t*)alloc((size_t)NROWS_X * CDIM * 2);
  u16t* cn_bf  = (u16t*)alloc((size_t)NROWS_C * CDIM * 2);
  u16t* Wq_bf  = (u16t*)alloc((size_t)CDIM * CDIM * 2);
  u16t* Wkv_bf = (u16t*)alloc((size_t)CDIM * 2 * CDIM * 2);
  u16t* Wo_bf  = (u16t*)alloc((size_t)CDIM * CDIM * 2);
  u16t* W1_bf  = (u16t*)alloc((size_t)CDIM * HID * 2);
  u16t* W2_bf  = (u16t*)alloc((size_t)HID * CDIM * 2);
  u16t* q_bf   = (u16t*)alloc((size_t)NROWS_X * CDIM * 2);
  u16t* kv_bf  = (u16t*)alloc((size_t)NROWS_C * 2 * CDIM * 2);
  u16t* o_bf   = (u16t*)alloc((size_t)NROWS_X * CDIM * 2);
  float* x1_f  = (float*)alloc((size_t)NROWS_X * CDIM * 4);
  u16t* xn2_bf = (u16t*)alloc((size_t)NROWS_X * CDIM * 2);
  u16t* h_bf   = (u16t*)alloc((size_t)NROWS_X * HID * 2);

  // 1) weight conversion f32 -> bf16
  auto cvt = [&](const float* src, u16t* dst, int n) {
    cvt_bf16_kernel<<<(n + 255) / 256, 256, 0, stream>>>(src, dst, n);
  };
  cvt(Wq,  Wq_bf,  CDIM * CDIM);
  cvt(Wkv, Wkv_bf, CDIM * 2 * CDIM);
  cvt(Wo,  Wo_bf,  CDIM * CDIM);
  cvt(W1,  W1_bf,  CDIM * HID);
  cvt(W2,  W2_bf,  HID * CDIM);

  // 2) layernorms (rows are multiples of 8 -> exact grids)
  ln_bf16_kernel<<<NROWS_X / 8, 256, 0, stream>>>(x, nullptr, nullptr, xn_bf,
                                                  1e-6f, NROWS_X);
  ln_bf16_kernel<<<NROWS_C / 8, 256, 0, stream>>>(ctx, g_ctx, b_ctx, cn_bf,
                                                  1e-5f, NROWS_C);

  // 3) Q = xn @ Wq + bq          (12288 x 384) @ (384 x 384)
  gemm_bf16_kernel<EPI_BF16>
      <<<dim3(CDIM / 128, NROWS_X / 128), 256, 0, stream>>>(
          xn_bf, Wq_bf, bq, nullptr, q_bf, nullptr, NROWS_X, CDIM, CDIM);

  // 4) KV = cn @ Wkv + bkv       (196608 x 384) @ (384 x 768)
  gemm_bf16_kernel<EPI_BF16>
      <<<dim3(2 * CDIM / 128, NROWS_C / 128), 256, 0, stream>>>(
          cn_bf, Wkv_bf, bkv, nullptr, kv_bf, nullptr, NROWS_C, 2 * CDIM, CDIM);

  // 5) attention -> o_bf
  attn_kernel<<<BT * NH, 128, 0, stream>>>(q_bf, kv_bf, o_bf);

  // 6) x1 = x + o @ Wo + bo      (fp32 out for next LN/residual)
  gemm_bf16_kernel<EPI_RESID_F32>
      <<<dim3(CDIM / 128, NROWS_X / 128), 256, 0, stream>>>(
          o_bf, Wo_bf, bo, x, nullptr, x1_f, NROWS_X, CDIM, CDIM);

  // 7) xn2 = LN(x1)
  ln_bf16_kernel<<<NROWS_X / 8, 256, 0, stream>>>(x1_f, nullptr, nullptr,
                                                  xn2_bf, 1e-6f, NROWS_X);

  // 8) h = gelu(xn2 @ W1 + b1)
  gemm_bf16_kernel<EPI_GELU_BF16>
      <<<dim3(HID / 128, NROWS_X / 128), 256, 0, stream>>>(
          xn2_bf, W1_bf, b1p, nullptr, h_bf, nullptr, NROWS_X, HID, CDIM);

  // 9) out = x1 + h @ W2 + b2
  gemm_bf16_kernel<EPI_RESID_F32>
      <<<dim3(CDIM / 128, NROWS_X / 128), 256, 0, stream>>>(
          h_bf, W2_bf, b2p, x1_f, nullptr, out, NROWS_X, CDIM, HID);
}
